// HistEncoder_27161373180245
// MI455X (gfx1250) — compile-verified, matrix operand order checked by reading, not run on hardware
//
#include <hip/hip_runtime.h>
#include <stdint.h>

// ---------------------------------------------------------------------------
// MI455X (gfx1250) implementation of the trajectory-transformer reference.
// Heavy GEMMs: v_wmma_f32_16x16x32_f16 (wave32 WMMA, f32 accumulate), with
// the shared W k-slice staged into LDS via global_load_async_to_lds_b128
// (ASYNCcnt DMA) and read back as B fragments via ds_load_b128.
// ---------------------------------------------------------------------------

typedef _Float16 half_t;
typedef __attribute__((ext_vector_type(16))) _Float16 v16h;
typedef __attribute__((ext_vector_type(8)))  _Float16 v8h;
typedef __attribute__((ext_vector_type(8)))  float     v8f;

namespace cfg {
constexpr int T   = 64;    // sequence length (S == T)
constexpr int B   = 32;    // batch
constexpr int E   = 64;    // embedding dim
constexpr int C   = 100;   // grid cells (G*G)
constexpr int N   = 3200;  // B*C neighbor count
constexpr int FD  = 6;     // raw feature dim
constexpr int FF  = 256;   // FFN hidden
constexpr int CTX = 128;   // 2*E
}

// ---------------------------------------------------------------------------
// Activation helper: 0 = none, 1 = relu, 2 = leaky(0.1)
// ---------------------------------------------------------------------------
__device__ __forceinline__ float apply_act(float x, int act) {
  if (act == 1) return x > 0.f ? x : 0.f;
  if (act == 2) return x > 0.f ? x : 0.1f * x;
  return x;
}

// ---------------------------------------------------------------------------
// WMMA GEMM: out[M,Nc] = act(A16[M,K] @ W16[Nc,K]^T + bias[Nc])
// A16, W16 row-major f16; K multiple of 32; M multiple of 64; Nc multiple 64.
//
// Block = 128 threads (4 waves) computing a 64(M) x 64(N) macro-tile:
//   wave w -> m-tile blockIdx.x*4+w, all 4 n-tiles of column block blockIdx.y.
// Per k-step the block stages the shared W k-slice (64 rows x 32 halves) into
// LDS with global_load_async_to_lds_b128 (2 x b128 per thread), waits on
// ASYNCcnt, then each wave loads one A fragment from global (b128 x2, with
// global_prefetch of the next chunk) and issues 4 WMMAs reading B fragments
// from LDS. LDS row stride 80B => lanes' b128 reads hit disjoint bank groups.
//
// Fragment layout per CDNA5 ISA 7.12.2 (16-bit A 16x32): lane r=lane%16 holds
// row r; h=lane/16 selects K-halves {h*8..h*8+7, 16+h*8..+7}; symmetric for B.
// ---------------------------------------------------------------------------
#define GEMM_WAVES 4
#define LDS_WROW   80   // padded LDS row stride (bytes) for a 64B k-slice row

__global__ __launch_bounds__(128) void wmma_gemm_kernel(
    const half_t* __restrict__ A, const half_t* __restrict__ W,
    const float* __restrict__ bias,
    float* __restrict__ outF, half_t* __restrict__ outH,
    int M, int K, int Nc, int act)
{
  __shared__ __align__(16) unsigned char lwbuf[64 * LDS_WROW];

  const int tid  = threadIdx.x;
  const int lane = tid & 31;
  const int wave = tid >> 5;
  const int mt    = blockIdx.x * GEMM_WAVES + wave;  // m-tile of this wave
  const int nbase = blockIdx.y * 64;                 // first of 4 n-tiles
  const int r = lane & 15;
  const int h = lane >> 4;

  const half_t* __restrict__ Arow = A + (size_t)(mt * 16 + r) * K + h * 8;

  // LDS byte address of the staging buffer (low 32 bits of the LDS aperture
  // address are the wave-relative LDS offset per ISA 10.2).
  const unsigned lbase = (unsigned)(size_t)(void*)lwbuf;

  // staging assignment: 256 x 16B chunks (64 rows x 4 parts), 2 per thread
  const int c0_row = tid >> 2,          c0_part = tid & 3;
  const int c1_row = (tid + 128) >> 2,  c1_part = tid & 3;

  v8f acc[4];
#pragma unroll
  for (int j = 0; j < 4; ++j) acc[j] = (v8f){};

  for (int k0 = 0; k0 < K; k0 += 32) {
    // ---- async-stage W[nbase + row][k0 .. k0+31] into LDS ----
    {
      const half_t* g0 = W + (size_t)(nbase + c0_row) * K + k0 + c0_part * 8;
      const half_t* g1 = W + (size_t)(nbase + c1_row) * K + k0 + c1_part * 8;
      unsigned l0 = lbase + (unsigned)(c0_row * LDS_WROW + c0_part * 16);
      unsigned l1 = lbase + (unsigned)(c1_row * LDS_WROW + c1_part * 16);
      asm volatile("global_load_async_to_lds_b128 %0, %1, off"
                   :: "v"(l0), "v"(g0) : "memory");
      asm volatile("global_load_async_to_lds_b128 %0, %1, off"
                   :: "v"(l1), "v"(g1) : "memory");
      asm volatile("s_wait_asynccnt 0" ::: "memory");
    }
    __syncthreads();   // all waves' DMA complete -> slice visible

    if (k0 + 32 < K)
      __builtin_prefetch((const void*)(Arow + k0 + 32), 0, 1);

    // ---- A fragment from global ----
    v8h a0 = *(const v8h*)(Arow + k0);
    v8h a1 = *(const v8h*)(Arow + k0 + 16);
    v16h a = __builtin_shufflevector(a0, a1,
               0,1,2,3,4,5,6,7,8,9,10,11,12,13,14,15);

    // ---- 4 B fragments from LDS, 4 WMMAs ----
    const unsigned char* lp = (const unsigned char*)lwbuf;
#pragma unroll
    for (int j = 0; j < 4; ++j) {
      const unsigned rb = (unsigned)((j * 16 + r) * LDS_WROW);
      v8h b0 = *(const v8h*)(lp + rb + h * 16);
      v8h b1 = *(const v8h*)(lp + rb + 32 + h * 16);
      v16h b = __builtin_shufflevector(b0, b1,
                 0,1,2,3,4,5,6,7,8,9,10,11,12,13,14,15);
      acc[j] = __builtin_amdgcn_wmma_f32_16x16x32_f16(
          /*neg_a=*/false, a, /*neg_b=*/false, b,
          /*c_mod=*/(short)0, acc[j], /*reuse_a=*/false, /*reuse_b=*/false);
    }
    __syncthreads();   // reads done before next k-step overwrites the slice
  }

#pragma unroll
  for (int j = 0; j < 4; ++j) {
    const int col = nbase + j * 16 + r;
    const float bcol = bias ? bias[col] : 0.f;
#pragma unroll
    for (int i = 0; i < 8; ++i) {
      float x = apply_act(acc[j][i] + bcol, act);
      size_t row = (size_t)(mt * 16) + i + 8 * h;  // D VGPR i -> M = i + 8*h
      size_t off = row * (size_t)Nc + col;
      if (outF) outF[off] = x;
      if (outH) outH[off] = (half_t)x;
    }
  }
}

static inline void gemm(hipStream_t s, const half_t* A, const half_t* W,
                        const float* bias, float* oF, half_t* oH,
                        int M, int K, int Nc, int act) {
  dim3 grid(M / 64, Nc / 64);
  wmma_gemm_kernel<<<grid, 128, 0, s>>>(A, W, bias, oF, oH, M, K, Nc, act);
}

// ---------------------------------------------------------------------------
// occupancy + scatter index (cumsum) for _scatter_grid. Mask bytes: any over L.
// ---------------------------------------------------------------------------
__global__ void occ_scan_kernel(const unsigned char* __restrict__ maskBytes,
                                int L, int Ngroups,
                                unsigned char* __restrict__ occ,
                                int* __restrict__ idx)
{
  for (int i = threadIdx.x; i < Ngroups; i += blockDim.x) {
    const unsigned char* p = maskBytes + (size_t)i * L;
    unsigned char any = 0;
    for (int k = 0; k < L; ++k) any |= p[k];
    occ[i] = any ? 1 : 0;
  }
  __syncthreads();
  if (threadIdx.x == 0) {
    int c = 0;
    for (int i = 0; i < Ngroups; ++i) {
      c += occ[i];
      int v = c - 1;
      if (v < 0) v = 0;
      if (v > Ngroups - 1) v = Ngroups - 1;
      idx[i] = v;
    }
  }
}

// ---------------------------------------------------------------------------
// Small-K linear (K=6): out[m,e] = act(sum_f src(m,f)*W[e,f] + b[e])
// rows m = t*Bq + n ; src(m,f) = raw[(row*T + t)*K + f] (* occ) where
// row = gidx ? gidx[n] : n.  Optionally writes f16.
// ---------------------------------------------------------------------------
__global__ void linear_small_k_kernel(
    const float* __restrict__ raw, const float* __restrict__ W,
    const float* __restrict__ bias,
    const int* __restrict__ gidx, const unsigned char* __restrict__ occ,
    float* __restrict__ outF, half_t* __restrict__ outH,
    int M, int T, int Bq, int K, int E, int act)
{
  int g = blockIdx.x * blockDim.x + threadIdx.x;
  if (g >= M * E) return;
  int e = g % E, m = g / E;
  int t = m / Bq, n = m % Bq;
  int srow = gidx ? gidx[n] : n;
  float msk = occ ? (occ[n] ? 1.f : 0.f) : 1.f;
  const float* rp = raw + ((size_t)srow * T + t) * K;
  float acc = bias[e];
  for (int f = 0; f < K; ++f) acc += (rp[f] * msk) * W[e * K + f];
  acc = apply_act(acc, act);
  if (outF) outF[(size_t)g] = acc;
  if (outH) outH[(size_t)g] = (half_t)acc;
}

// ---------------------------------------------------------------------------
// elementwise helpers
// ---------------------------------------------------------------------------
__global__ void f32_to_f16_kernel(const float* __restrict__ a,
                                  half_t* __restrict__ o, int n) {
  int g = blockIdx.x * blockDim.x + threadIdx.x;
  if (g < n) o[g] = (half_t)a[g];
}

__global__ void leaky_kernel(const float* __restrict__ a,
                             float* __restrict__ o, int n) {
  int g = blockIdx.x * blockDim.x + threadIdx.x;
  if (g < n) { float v = a[g]; o[g] = v > 0.f ? v : 0.1f * v; }
}

// ---------------------------------------------------------------------------
// q16 = f16(x + pos_encoding(raw)) ; pos computed inline.
// x rows m = t*Bq + b ; raw coords raw[(b*T+t)*FD + {0,1}].
// pos[e]: e<32 from y-coord, e>=32 from x-coord; within a 32-block:
//   even -> sin, odd -> cos ; dim_t = 10000^(2*floor(e'/2)/32).
// ---------------------------------------------------------------------------
__global__ void make_q16_kernel(const float* __restrict__ x,
                                const float* __restrict__ raw,
                                half_t* __restrict__ q16,
                                int M, int T, int Bq, int FDc)
{
  int g = blockIdx.x * blockDim.x + threadIdx.x;
  if (g >= M * 64) return;
  int e = g & 63, m = g >> 6;
  int t = m / Bq, b = m % Bq;
  int blockSel = e >> 5;          // 0: py (coord 1), 1: px (coord 0)
  int ep = e & 31;
  float coord = raw[((size_t)b * T + t) * FDc + (blockSel == 0 ? 1 : 0)];
  float dimt = __powf(10000.f, (2.f * (float)(ep >> 1)) * (1.f / 32.f));
  float p = coord * 6.283185307f / dimt;
  float pos = (ep & 1) ? __cosf(p) : __sinf(p);
  q16[(size_t)g] = (half_t)(x[(size_t)g] + pos);
}

// ---------------------------------------------------------------------------
// Encoder MHA core. QK[m,0:64]=Q, [64:128]=K (m = s*Bq+b), V[m,0:64].
// One thread per (s,b,h); streaming 3-pass softmax over t (S=64, hd=16).
// Writes f16 (S,Bq,64).
// ---------------------------------------------------------------------------
__global__ void mha_core_kernel(const float* __restrict__ QK,
                                const float* __restrict__ V,
                                half_t* __restrict__ outH,
                                int S, int Bq)
{
  int gid = blockIdx.x * blockDim.x + threadIdx.x;
  int total = S * Bq * 4;
  if (gid >= total) return;
  int hh = gid & 3;
  int tmp = gid >> 2;
  int b = tmp % Bq, s = tmp / Bq;

  float q[16];
  const float* Qp = QK + ((size_t)s * Bq + b) * 128 + hh * 16;
#pragma unroll
  for (int d = 0; d < 16; ++d) q[d] = Qp[d] * 0.25f;   // 1/sqrt(16)

  float mx = -1e30f;
  for (int t = 0; t < S; ++t) {
    const float* Kp = QK + ((size_t)t * Bq + b) * 128 + 64 + hh * 16;
    float sdot = 0.f;
#pragma unroll
    for (int d = 0; d < 16; ++d) sdot += q[d] * Kp[d];
    mx = fmaxf(mx, sdot);
  }
  float denom = 0.f;
  for (int t = 0; t < S; ++t) {
    const float* Kp = QK + ((size_t)t * Bq + b) * 128 + 64 + hh * 16;
    float sdot = 0.f;
#pragma unroll
    for (int d = 0; d < 16; ++d) sdot += q[d] * Kp[d];
    denom += __expf(sdot - mx);
  }
  float o[16];
#pragma unroll
  for (int d = 0; d < 16; ++d) o[d] = 0.f;
  for (int t = 0; t < S; ++t) {
    const float* Kp = QK + ((size_t)t * Bq + b) * 128 + 64 + hh * 16;
    float sdot = 0.f;
#pragma unroll
    for (int d = 0; d < 16; ++d) sdot += q[d] * Kp[d];
    float w = __expf(sdot - mx);
    const float* Vp = V + ((size_t)t * Bq + b) * 64 + hh * 16;
#pragma unroll
    for (int d = 0; d < 16; ++d) o[d] += w * Vp[d];
  }
  half_t* op = outH + ((size_t)s * Bq + b) * 64 + hh * 16;
  float inv = 1.f / denom;
#pragma unroll
  for (int d = 0; d < 16; ++d) op[d] = (half_t)(o[d] * inv);
}

// ---------------------------------------------------------------------------
// Fused residual-add + LayerNorm over E=64; one 64-thread block per row.
// transA: a-row index = (r%T)*Bq + r/T (for the (T,B)->(B,T) transpose add).
// ---------------------------------------------------------------------------
__global__ __launch_bounds__(64) void add_ln_kernel(
    const float* __restrict__ a, const float* __restrict__ res,
    const float* __restrict__ g, const float* __restrict__ beta,
    float* __restrict__ yF, half_t* __restrict__ yH,
    int T, int Bq, int transA)
{
  int r = blockIdx.x, e = threadIdx.x;
  size_t arow = (size_t)r;
  if (transA) { int t = r % T; int b = r / T; arow = (size_t)t * Bq + b; }
  float v = a[arow * 64 + e] + res[(size_t)r * 64 + e];

  __shared__ float red[64];
  __shared__ float s_mean, s_rstd;
  red[e] = v; __syncthreads();
  for (int s = 32; s > 0; s >>= 1) { if (e < s) red[e] += red[e + s]; __syncthreads(); }
  if (e == 0) s_mean = red[0] * (1.f / 64.f);
  __syncthreads();
  float d = v - s_mean;
  red[e] = d * d; __syncthreads();
  for (int s = 32; s > 0; s >>= 1) { if (e < s) red[e] += red[e + s]; __syncthreads(); }
  if (e == 0) s_rstd = rsqrtf(red[0] * (1.f / 64.f) + 1e-5f);
  __syncthreads();
  float y = d * s_rstd * g[e] + beta[e];
  yF[(size_t)r * 64 + e] = y;
  if (yH) yH[(size_t)r * 64 + e] = (half_t)y;
}

// ---------------------------------------------------------------------------
// Grid attention (C cells, masked softmax + renormalize, per reference).
// Q rows (t*BQ+b), K/V rows (t*BQ*C + b*C + c), out rows (b*T+t).
// One thread per (t,b,h); D=16.
// ---------------------------------------------------------------------------
__global__ void grid_attn_kernel(const float* __restrict__ Q,
                                 const float* __restrict__ Kx,
                                 const float* __restrict__ Vx,
                                 const unsigned char* __restrict__ occ,
                                 float* __restrict__ out,
                                 int T, int BQ, int C)
{
  int gid = blockIdx.x * blockDim.x + threadIdx.x;
  int total = T * BQ * 4;
  if (gid >= total) return;
  int hh = gid & 3;
  int tmp = gid >> 2;
  int b = tmp % BQ, t = tmp / BQ;
  const int NKV = BQ * C;

  float q[16];
  const float* Qp = Q + ((size_t)t * BQ + b) * 64 + hh * 16;
#pragma unroll
  for (int d = 0; d < 16; ++d) q[d] = Qp[d] * 0.25f;   // 1/sqrt(D=16)

  float mx = -1e30f;
  for (int c = 0; c < C; ++c) {
    float s = -1e9f;
    if (occ[b * C + c]) {
      const float* Kp = Kx + ((size_t)t * NKV + b * C + c) * 64 + hh * 16;
      float sdot = 0.f;
#pragma unroll
      for (int d = 0; d < 16; ++d) sdot += q[d] * Kp[d];
      s = sdot;
    }
    mx = fmaxf(mx, s);
  }
  float denom1 = 0.f;
  for (int c = 0; c < C; ++c) {
    float s = -1e9f;
    if (occ[b * C + c]) {
      const float* Kp = Kx + ((size_t)t * NKV + b * C + c) * 64 + hh * 16;
      float sdot = 0.f;
#pragma unroll
      for (int d = 0; d < 16; ++d) sdot += q[d] * Kp[d];
      s = sdot;
    }
    denom1 += __expf(s - mx);
  }
  float num[16];
#pragma unroll
  for (int d = 0; d < 16; ++d) num[d] = 0.f;
  float denom2 = 1e-6f;
  float inv1 = 1.f / denom1;
  for (int c = 0; c < C; ++c) {
    if (!occ[b * C + c]) continue;
    const float* Kp = Kx + ((size_t)t * NKV + b * C + c) * 64 + hh * 16;
    float sdot = 0.f;
#pragma unroll
    for (int d = 0; d < 16; ++d) sdot += q[d] * Kp[d];
    float w = __expf(sdot - mx) * inv1;
    denom2 += w;
    const float* Vp = Vx + ((size_t)t * NKV + b * C + c) * 64 + hh * 16;
#pragma unroll
    for (int d = 0; d < 16; ++d) num[d] += w * Vp[d];
  }
  float* op = out + ((size_t)b * T + t) * 64 + hh * 16;
  float inv2 = 1.f / denom2;
#pragma unroll
  for (int d = 0; d < 16; ++d) op[d] = num[d] * inv2;
}

// ---------------------------------------------------------------------------
// soc16[(t*N+n)*64+e] = f16(x[(t*N+idx[n])*64+e] * occ[n])
// ---------------------------------------------------------------------------
__global__ void gather_f16_kernel(const float* __restrict__ x,
                                  const int* __restrict__ idx,
                                  const unsigned char* __restrict__ occ,
                                  half_t* __restrict__ out, int T, int N)
{
  int g = blockIdx.x * blockDim.x + threadIdx.x;
  if (g >= T * N * 64) return;
  int e = g & 63;
  int mm = g >> 6;
  int n = mm % N, t = mm / N;
  float v = occ[n] ? x[((size_t)t * N + idx[n]) * 64 + e] : 0.f;
  out[(size_t)g] = (half_t)v;
}

// ---------------------------------------------------------------------------
// cross[b,t,0:64]=hist[t,b,:]; cross[b,t,64:128]=leaky(tval+sval)
// ---------------------------------------------------------------------------
__global__ void finalize_cross_kernel(const float* __restrict__ hist,
                                      const float* __restrict__ tval,
                                      const float* __restrict__ sval,
                                      float* __restrict__ out)
{
  int g = blockIdx.x * blockDim.x + threadIdx.x;
  if (g >= cfg::B * cfg::T * cfg::CTX) return;
  int e = g & 127;
  int m = g >> 7;              // m = b*T + t
  int t = m % cfg::T, b = m / cfg::T;
  float v;
  if (e < 64) {
    v = hist[((size_t)t * cfg::B + b) * 64 + e];
  } else {
    float a = tval[(size_t)m * 64 + (e - 64)] + sval[(size_t)m * 64 + (e - 64)];
    v = a > 0.f ? a : 0.1f * a;
  }
  out[(size_t)g] = v;
}

// ---------------------------------------------------------------------------
// Global head: mean over T -> LN(gln) -> g1 GEMV -> exact GELU -> g2 GEMV.
// One 128-thread block per batch element.
// ---------------------------------------------------------------------------
__global__ __launch_bounds__(128) void global_head_kernel(
    const float* __restrict__ cross,
    const float* __restrict__ glg, const float* __restrict__ glb,
    const float* __restrict__ g1w, const float* __restrict__ g1b,
    const float* __restrict__ g2w, const float* __restrict__ g2b,
    float* __restrict__ gout)
{
  int b = blockIdx.x, j = threadIdx.x;   // 128 threads
  float m = 0.f;
  for (int t = 0; t < cfg::T; ++t)
    m += cross[((size_t)b * cfg::T + t) * cfg::CTX + j];
  m *= (1.f / (float)cfg::T);

  __shared__ float sh[128];
  __shared__ float red[128];
  __shared__ float s_mean, s_rstd;
  red[j] = m; __syncthreads();
  for (int s = 64; s > 0; s >>= 1) { if (j < s) red[j] += red[j + s]; __syncthreads(); }
  if (j == 0) s_mean = red[0] * (1.f / 128.f);
  __syncthreads();
  float d = m - s_mean;
  red[j] = d * d; __syncthreads();
  for (int s = 64; s > 0; s >>= 1) { if (j < s) red[j] += red[j + s]; __syncthreads(); }
  if (j == 0) s_rstd = rsqrtf(red[0] * (1.f / 128.f) + 1e-5f);
  __syncthreads();
  sh[j] = d * s_rstd * glg[j] + glb[j];
  __syncthreads();

  float acc = g1b[j];
  for (int k = 0; k < 128; ++k) acc += sh[k] * g1w[j * 128 + k];
  float ge = 0.5f * acc * (1.f + erff(acc * 0.70710678f));   // exact GELU
  __syncthreads();
  sh[j] = ge;
  __syncthreads();
  float acc2 = g2b[j];
  for (int k = 0; k < 128; ++k) acc2 += sh[k] * g2w[j * 128 + k];
  gout[(size_t)b * 128 + j] = acc2;
}

// ---------------------------------------------------------------------------
// Host orchestration
// ---------------------------------------------------------------------------
extern "C" void kernel_launch(void* const* d_in, const int* in_sizes, int n_in,
                              void* d_out, int out_size, void* d_ws, size_t ws_size,
                              hipStream_t stream) {
  using namespace cfg;
  (void)in_sizes; (void)n_in; (void)out_size; (void)ws_size;

  const float* src  = (const float*)d_in[0];            // (B,T,FD)
  const float* nbrs = (const float*)d_in[1];            // (N,T,FD)
  const unsigned char* maskS = (const unsigned char*)d_in[2];  // (B,C,E) bool
  const unsigned char* maskT = (const unsigned char*)d_in[3];  // (B,C,FD) bool
  auto P = [&](int i) { return (const float*)d_in[i]; };

  const float* emb_w = P(4);  const float* emb_b = P(5);
  const float* tq_w = P(6);   const float* tq_b = P(7);
  const float* tk_w = P(8);   const float* tk_b = P(9);
  const float* tv_w = P(10);  const float* tv_b = P(11);
  const float* qf_w = P(12);  const float* qf_b = P(13);
  const float* kf_w = P(14);  const float* kf_b = P(15);
  const float* vf_w = P(16);  const float* vf_b = P(17);
  const float* res_g = P(18); const float* res_b = P(19);
  // layers at 20.. and 32..: ipw,ipb,opw,opb,l1w,l1b,l2w,l2b,n1g,n1b,n2g,n2b
  const int LBASE[2] = {20, 32};
  const float* gln_g = P(44); const float* gln_b = P(45);
  const float* g1_w = P(46);  const float* g1_b = P(47);
  const float* g2_w = P(48);  const float* g2_b = P(49);

  // -------- workspace bump allocator (256B aligned regions) ----------------
  char* ws = (char*)d_ws;
  size_t wo = 0;
  auto walloc = [&](size_t bytes) -> void* {
    void* p = ws + wo;
    wo = (wo + bytes + 255) & ~(size_t)255;
    return p;
  };

  const int M1 = T * B;      // 2048
  const int M2 = T * N;      // 204800

  // f16 weights
  half_t* ipw16[2]; half_t* opw16[2]; half_t* l1w16[2]; half_t* l2w16[2];
  for (int l = 0; l < 2; ++l) {
    ipw16[l] = (half_t*)walloc(192 * 64 * sizeof(half_t));
    opw16[l] = (half_t*)walloc(64 * 64 * sizeof(half_t));
    l1w16[l] = (half_t*)walloc(256 * 64 * sizeof(half_t));
    l2w16[l] = (half_t*)walloc(64 * 256 * sizeof(half_t));
  }
  half_t* qf16 = (half_t*)walloc(64 * 64 * sizeof(half_t));
  half_t* kf16 = (half_t*)walloc(64 * 64 * sizeof(half_t));
  half_t* vf16 = (half_t*)walloc(64 * 64 * sizeof(half_t));

  // small buffers
  int* idx_t = (int*)walloc(N * sizeof(int));
  int* idx_s = (int*)walloc(N * sizeof(int));
  unsigned char* occ_t = (unsigned char*)walloc(N);
  unsigned char* occ_s = (unsigned char*)walloc(N);
  float* tval   = (float*)walloc((size_t)B * T * 64 * 4);
  float* sval   = (float*)walloc((size_t)B * T * 64 * 4);
  float* embsrc = (float*)walloc((size_t)M1 * 64 * 4);
  float* histK  = (float*)walloc((size_t)M1 * 64 * 4);
  half_t* hist16 = (half_t*)walloc((size_t)M1 * 64 * 2);
  float* Qs     = (float*)walloc((size_t)M1 * 64 * 4);
  float* tQ     = (float*)walloc((size_t)M1 * 64 * 4);

  // big regions (sized for M2, reused across phases)
  float*  X    = (float*)walloc((size_t)M2 * 64 * 4);    // encoder activations
  half_t* Q16  = (half_t*)walloc((size_t)M2 * 64 * 2);   // x+pos f16
  half_t* X16  = (half_t*)walloc((size_t)M2 * 64 * 2);   // x f16
  half_t* MH16 = (half_t*)walloc((size_t)M2 * 64 * 2);   // mha out / soc16
  float*  Vbuf = (float*)walloc((size_t)M2 * 64 * 4);    // V / proj / ffn2 / Ks
  float*  QKbuf= (float*)walloc((size_t)M2 * 128 * 4);   // QK / tK+tV / Vs
  half_t* F1   = (half_t*)walloc((size_t)M2 * 256 * 2);  // ffn hidden f16

  auto cvt = [&](const float* a, half_t* o, int n) {
    f32_to_f16_kernel<<<(n + 255) / 256, 256, 0, stream>>>(a, o, n);
  };

  // -------- phase 0: weight conversion -------------------------------------
  for (int l = 0; l < 2; ++l) {
    cvt(P(LBASE[l] + 0), ipw16[l], 192 * 64);
    cvt(P(LBASE[l] + 2), opw16[l], 64 * 64);
    cvt(P(LBASE[l] + 4), l1w16[l], 256 * 64);
    cvt(P(LBASE[l] + 6), l2w16[l], 64 * 256);
  }
  cvt(qf_w, qf16, 64 * 64);
  cvt(kf_w, kf16, 64 * 64);
  cvt(vf_w, vf16, 64 * 64);

  // -------- phase A: temporal grid attention -------------------------------
  occ_scan_kernel<<<1, 256, 0, stream>>>(maskT, FD, N, occ_t, idx_t);
  occ_scan_kernel<<<1, 256, 0, stream>>>(maskS, E, N, occ_s, idx_s);

  float* tK = QKbuf;
  float* tV = QKbuf + (size_t)M2 * 64;
  {
    int n1 = M1 * 64, n2 = M2 * 64;
    linear_small_k_kernel<<<(n1 + 255) / 256, 256, 0, stream>>>(
        src, tq_w, tq_b, nullptr, nullptr, tQ, nullptr, M1, T, B, FD, 64, 0);
    linear_small_k_kernel<<<(n2 + 255) / 256, 256, 0, stream>>>(
        nbrs, tk_w, tk_b, idx_t, occ_t, tK, nullptr, M2, T, N, FD, 64, 0);
    linear_small_k_kernel<<<(n2 + 255) / 256, 256, 0, stream>>>(
        nbrs, tv_w, tv_b, idx_t, occ_t, tV, nullptr, M2, T, N, FD, 64, 0);
  }
  grid_attn_kernel<<<(T * B * 4 + 255) / 256, 256, 0, stream>>>(
      tQ, tK, tV, occ_t, tval, T, B, C);

  // emb_src and hist-encoder input
  {
    int n1 = M1 * 64;
    linear_small_k_kernel<<<(n1 + 255) / 256, 256, 0, stream>>>(
        src, emb_w, emb_b, nullptr, nullptr, embsrc, nullptr, M1, T, B, FD, 64, 0);
    leaky_kernel<<<(n1 + 255) / 256, 256, 0, stream>>>(embsrc, X, n1);
  }
  // tval = LN(emb_src^T + tval) with res_g/res_b   (transposed add)
  add_ln_kernel<<<M1, 64, 0, stream>>>(embsrc, tval, res_g, res_b,
                                       tval, nullptr, T, B, 1);

  // -------- shared encoder runner ------------------------------------------
  auto run_encoder = [&](int Bq, const float* raw) {
    int M = T * Bq;
    int nE = M * 64;
    for (int l = 0; l < 2; ++l) {
      const float* ipb = P(LBASE[l] + 1);
      const float* opb = P(LBASE[l] + 3);
      const float* l1b = P(LBASE[l] + 5);
      const float* l2b = P(LBASE[l] + 7);
      const float* n1g = P(LBASE[l] + 8);
      const float* n1b = P(LBASE[l] + 9);
      const float* n2g = P(LBASE[l] + 10);
      const float* n2b = P(LBASE[l] + 11);

      make_q16_kernel<<<(nE + 255) / 256, 256, 0, stream>>>(X, raw, Q16, M, T, Bq, FD);
      cvt(X, X16, nE);
      // QK = q @ [wq;wk]^T + b  (first 128 rows of ipw)
      gemm(stream, Q16, ipw16[l], ipb, QKbuf, nullptr, M, 64, 128, 0);
      // V = x @ wv^T + bv       (rows 128..191 of ipw)
      gemm(stream, X16, ipw16[l] + 128 * 64, ipb + 128, Vbuf, nullptr, M, 64, 64, 0);
      mha_core_kernel<<<(M * 4 + 255) / 256, 256, 0, stream>>>(QKbuf, Vbuf, MH16, T, Bq);
      // attn proj (overwrites Vbuf)
      gemm(stream, MH16, opw16[l], opb, Vbuf, nullptr, M, 64, 64, 0);
      add_ln_kernel<<<M, 64, 0, stream>>>(X, Vbuf, n1g, n1b, X, X16, T, Bq, 0);
      // FFN
      gemm(stream, X16, l1w16[l], l1b, nullptr, F1, M, 64, 256, 1);   // relu, f16 out
      gemm(stream, F1, l2w16[l], l2b, Vbuf, nullptr, M, 256, 64, 0);
      add_ln_kernel<<<M, 64, 0, stream>>>(X, Vbuf, n2g, n2b, X, nullptr, T, Bq, 0);
    }
  };

  // -------- phase B: history encoder (M=2048) ------------------------------
  run_encoder(B, src);
  hipMemcpyAsync(histK, X, (size_t)M1 * 64 * 4, hipMemcpyDeviceToDevice, stream);
  cvt(X, hist16, M1 * 64);

  // -------- phase C: neighbor encoder (M=204800) ---------------------------
  {
    int n2 = M2 * 64;
    linear_small_k_kernel<<<(n2 + 255) / 256, 256, 0, stream>>>(
        nbrs, emb_w, emb_b, nullptr, nullptr, X, nullptr, M2, T, N, FD, 64, 2); // leaky
  }
  run_encoder(N, nbrs);

  // -------- phase D: spatial grid attention + outputs ----------------------
  half_t* soc16 = MH16;
  gather_f16_kernel<<<(M2 * 64 + 255) / 256, 256, 0, stream>>>(
      X, idx_s, occ_s, soc16, T, N);
  gemm(stream, hist16, qf16, qf_b, Qs, nullptr, M1, 64, 64, 0);
  gemm(stream, soc16, kf16, kf_b, Vbuf, nullptr, M2, 64, 64, 0);    // Ks
  gemm(stream, soc16, vf16, vf_b, QKbuf, nullptr, M2, 64, 64, 0);   // Vs
  grid_attn_kernel<<<(T * B * 4 + 255) / 256, 256, 0, stream>>>(
      Qs, Vbuf, QKbuf, occ_s, sval, T, B, C);

  float* outF = (float*)d_out;
  finalize_cross_kernel<<<(B * T * CTX + 255) / 256, 256, 0, stream>>>(
      histK, tval, sval, outF);
  global_head_kernel<<<B, 128, 0, stream>>>(
      outF, gln_g, gln_b, g1_w, g1_b, g2_w, g2_b, outF + (size_t)B * T * CTX);
}
